// Decoder_64922725646612
// MI455X (gfx1250) — compile-verified, matrix operand order checked by reading, not run on hardware
//
#include <hip/hip_runtime.h>
#include <hip/hip_bf16.h>

// ---------------------------------------------------------------------------
// 4-layer LSTM decoder for MI455X (gfx1250).
//  * bf16 WMMA (v_wmma_f32_16x16x32_bf16) with fp32 accumulate; weights are
//    converted once to bf16 (59 MB) so the whole recurrence streams from the
//    192 MB L2, not HBM.
//  * ONE persistent kernel runs all 127 steps: 66 workgroups (48 hgates +
//    16 layer-0-cell + 2 projection tasks), agent-scope atomic grid barrier
//    between dependency phases (4 barriers/step instead of 6 kernel launches).
//    66 WGs x 50KB LDS is trivially co-resident on a >=128-WGP part.
//  * K-chunk staging uses CDNA5 GLOBAL_LOAD_ASYNC_TO_LDS_B128 (ASYNCcnt path)
//    when the builtin exists, falling back to global->VGPR->ds_store.
// ---------------------------------------------------------------------------

typedef __bf16 bf16_t;
typedef __attribute__((ext_vector_type(16))) __bf16 v16bf;
typedef __attribute__((ext_vector_type(8)))  __bf16 v8bf;
typedef __attribute__((ext_vector_type(8)))  float  v8f;
typedef __attribute__((ext_vector_type(4)))  int    v4i;

#define NHID   1024
#define NOUT   512
#define BSZ    64
#define NLAY   4
#define STEPS  128
#define BK     32            // K chunk (one wmma K depth)
#define NK     (NHID / BK)   // 32 K chunks
#define AP     40            // padded LDS row stride (bf16) -> 80B, 16B aligned
#define BP     40
#define HL     (BSZ * NHID)            // per-layer h/c elems   (65536)
#define WL     ((long)4 * NHID * NHID) // per-layer weight elems (4194304)
#define GL     ((long)BSZ * 4 * NHID)  // per-layer gate elems  (262144)
#define NWG    66

#if defined(__has_builtin)
#  if __has_builtin(__builtin_amdgcn_global_load_async_to_lds_b128) && \
      __has_builtin(__builtin_amdgcn_s_wait_asynccnt)
#    define HAVE_ASYNC 1
#  endif
#endif
#ifndef HAVE_ASYNC
#  define HAVE_ASYNC 0
#endif

#if HAVE_ASYNC
// Probe-learned signature: (int4 AS1* gsrc, int4 AS3* ldst, imm offset, imm cpol)
typedef __attribute__((address_space(1))) v4i g_v4i;
typedef __attribute__((address_space(3))) v4i l_v4i;
#define ASYNC_B128(gsrc, ldst)                                                \
    __builtin_amdgcn_global_load_async_to_lds_b128(                           \
        (g_v4i*)(gsrc), (l_v4i*)(ldst), 0, 0)
#define ASYNC_WAIT() __builtin_amdgcn_s_wait_asynccnt(0)
#endif

__device__ __forceinline__ float sig_f(float x) {
    return 1.0f / (1.0f + __expf(-x));
}
__device__ __forceinline__ float tanh_f(float x) {
    x = fminf(fmaxf(x, -15.0f), 15.0f);
    float e = __expf(2.0f * x);
    return (e - 1.0f) / (e + 1.0f);
}

// Monotonic grid barrier: release add, acquire spin (s_sleep between polls).
__device__ __forceinline__ void grid_bar(unsigned* cnt, unsigned target) {
    __syncthreads();
    if (threadIdx.x == 0) {
        __hip_atomic_fetch_add(cnt, 1u, __ATOMIC_RELEASE, __HIP_MEMORY_SCOPE_AGENT);
        while (__hip_atomic_load(cnt, __ATOMIC_ACQUIRE, __HIP_MEMORY_SCOPE_AGENT) < target)
            __builtin_amdgcn_s_sleep(1);
    }
    __syncthreads();
}

// ---------------------------------------------------------------------------
// One WG-tile GEMM: gates[b, col] = sum_k A[b,k]*B[row(col),k],
//   col = colBase + gblk*cbs + nLocal (4 blocks of 16-col tiles per wave).
// hOut != nullptr -> fused LSTM cell epilogue; else raw fp32 store.
// ---------------------------------------------------------------------------
__device__ __forceinline__ void gemm_tile(
    const bf16_t* __restrict__ A, const bf16_t* __restrict__ B,
    const float*  __restrict__ addend,
    float*        __restrict__ gatesOut, int outW,
    const float*  __restrict__ cIn, float* __restrict__ cOut,
    bf16_t*       __restrict__ hOut,
    int colBase, int cbs,
    bf16_t* ldsA, bf16_t* ldsB)
{
    const int t    = threadIdx.x;
    const int wave = t >> 5, lane = t & 31;
    const int mw   = wave >> 2;        // 0..1 : 32-row half of the batch
    const int nw   = wave & 3;         // 0..3 : 16-col tile within each gate blk

    // staging assignment (one b128 for A, four for B, per thread per chunk)
    const int  aRow = t >> 2;
    const int  aSeg = (t & 3) << 3;
    const long aBase = (long)aRow * NHID + aSeg;
    const int  bRowGlobal = colBase + (t >> 6) * cbs + (t & 63);
    const long bBase = (long)bRowGlobal * NHID;
    bf16_t* lA = ldsA + aRow * AP + aSeg;
    bf16_t* lB = ldsB + t * BP;

#if HAVE_ASYNC
    // prologue: async-stage chunk 0
    ASYNC_B128(A + aBase, lA);
#pragma unroll
    for (int s = 0; s < 4; ++s) ASYNC_B128(B + bBase + s * 8, lB + s * 8);
    ASYNC_WAIT();
    __syncthreads();
#else
    v8bf stA = *(const v8bf*)(A + aBase);
    v8bf stB[4];
#pragma unroll
    for (int s = 0; s < 4; ++s) stB[s] = *(const v8bf*)(B + bBase + s * 8);
    *(v8bf*)lA = stA;
#pragma unroll
    for (int s = 0; s < 4; ++s) *(v8bf*)(lB + s * 8) = stB[s];
    __syncthreads();
#endif

    v8f acc[2][4];
#pragma unroll
    for (int mi = 0; mi < 2; ++mi)
#pragma unroll
        for (int g = 0; g < 4; ++g) { v8f z = {}; acc[mi][g] = z; }

    const int mLane = lane & 15;
    const int kh    = (lane >> 4) << 3;

    for (int kc = 0; kc < NK; ++kc) {
        const int cur = kc & 1;

        // stage next chunk while computing this one
        if (kc + 1 < NK) {
            const long ko  = (long)(kc + 1) * BK;
            const int  nxt = cur ^ 1;
#if HAVE_ASYNC
            ASYNC_B128(A + aBase + ko, lA + nxt * (64 * AP));
#pragma unroll
            for (int s = 0; s < 4; ++s)
                ASYNC_B128(B + bBase + ko + s * 8, lB + nxt * (256 * BP) + s * 8);
#else
            stA = *(const v8bf*)(A + aBase + ko);
#pragma unroll
            for (int s = 0; s < 4; ++s)
                stB[s] = *(const v8bf*)(B + bBase + ko + s * 8);
            (void)nxt;
#endif
        }

        // fragments: K-pairs along VGPRs, M/N along lanes, low/high lane
        // halves carry k 0-7,16-23 / k 8-15,24-31 (per ISA 16-bit layouts).
        v16bf aF[2], bF[4];
#pragma unroll
        for (int mi = 0; mi < 2; ++mi) {
            const int m = mw * 32 + mi * 16 + mLane;
            const bf16_t* p = ldsA + cur * (64 * AP) + m * AP + kh;
            v8bf lo = *(const v8bf*)p;
            v8bf hi = *(const v8bf*)(p + 16);
#pragma unroll
            for (int i = 0; i < 8; ++i) { aF[mi][i] = lo[i]; aF[mi][8 + i] = hi[i]; }
        }
#pragma unroll
        for (int g = 0; g < 4; ++g) {
            const int n = g * 64 + nw * 16 + mLane;
            const bf16_t* p = ldsB + cur * (256 * BP) + n * BP + kh;
            v8bf lo = *(const v8bf*)p;
            v8bf hi = *(const v8bf*)(p + 16);
#pragma unroll
            for (int i = 0; i < 8; ++i) { bF[g][i] = lo[i]; bF[g][8 + i] = hi[i]; }
        }

#pragma unroll
        for (int mi = 0; mi < 2; ++mi)
#pragma unroll
            for (int g = 0; g < 4; ++g)
                acc[mi][g] = __builtin_amdgcn_wmma_f32_16x16x32_bf16(
                    false, aF[mi], false, bF[g], (short)0, acc[mi][g], false, false);

        if (kc + 1 < NK) {
#if !HAVE_ASYNC
            const int nxt = cur ^ 1;
            *(v8bf*)(lA + nxt * (64 * AP)) = stA;
#pragma unroll
            for (int s = 0; s < 4; ++s) *(v8bf*)(lB + nxt * (256 * BP) + s * 8) = stB[s];
#endif
        }
#if HAVE_ASYNC
        ASYNC_WAIT();
#endif
        __syncthreads();
    }

    // ---- epilogue ----------------------------------------------------------
    if (hOut) {
        // fused LSTM cell: each lane owns matching i/f/c/o values in-register
#pragma unroll
        for (int mi = 0; mi < 2; ++mi) {
#pragma unroll
            for (int r = 0; r < 8; ++r) {
                const int b  = mw * 32 + mi * 16 + r + ((lane >> 4) << 3);
                const int hc = colBase + nw * 16 + (lane & 15);
                float gi = acc[mi][0][r], gf = acc[mi][1][r];
                float gc = acc[mi][2][r], go = acc[mi][3][r];
                if (addend) {
                    const float* ad = addend + (long)b * 4096 + hc;
                    gi += ad[0]; gf += ad[1024]; gc += ad[2048]; go += ad[3072];
                }
                const float i_ = sig_f(gi), f_ = sig_f(gf);
                const float c_ = tanh_f(gc), o_ = sig_f(go);
                const long  idx = (long)b * NHID + hc;
                const float cy  = f_ * cIn[idx] + i_ * c_;
                cOut[idx] = cy;
                hOut[idx] = (bf16_t)(o_ * tanh_f(cy));
            }
        }
    } else {
#pragma unroll
        for (int mi = 0; mi < 2; ++mi)
#pragma unroll
            for (int g = 0; g < 4; ++g)
#pragma unroll
                for (int r = 0; r < 8; ++r) {
                    const int b   = mw * 32 + mi * 16 + r + ((lane >> 4) << 3);
                    const int col = colBase + g * cbs + nw * 16 + (lane & 15);
                    gatesOut[(long)b * outW + col] = acc[mi][g][r];
                }
    }
}

// ---------------------------------------------------------------------------
// Persistent recurrence kernel: 66 WGs, 127 steps, 4 grid barriers per step.
//   Phase A: WG  0..47  hgates layers 1..3      (hPrev -> gb)
//            WG 48..63  layer-0 GEMM + cell     (hPrev[0],c0 -> hNext[0],c0)
//            WG 64..65  out[t-1] = hPrev[3]@L^T (free: phase A never writes hPrev)
//   Phase B/C/D: WG 0..15 layer l igates GEMM + stored hgates + cell.
// ---------------------------------------------------------------------------
__global__ __launch_bounds__(256)
void lstm_persistent(const bf16_t* __restrict__ Wbf, const bf16_t* __restrict__ Ubf,
                     const bf16_t* __restrict__ Lbf,
                     bf16_t* __restrict__ hA, bf16_t* __restrict__ hB,
                     float* __restrict__ c, float* __restrict__ gb,
                     float* __restrict__ out, unsigned* __restrict__ bar)
{
    __shared__ bf16_t ldsA[2 * 64 * AP];
    __shared__ bf16_t ldsB[2 * 256 * BP];

    const int w = blockIdx.x;
    unsigned epoch = 0;

    bf16_t* hPrev = hA;
    bf16_t* hNext = hB;
    for (int t = 1; t < STEPS; ++t) {
        // ---- Phase A --------------------------------------------------------
        if (w < 48) {
            const int l = 1 + (w >> 4), s = w & 15;
            gemm_tile(hPrev + (size_t)l * HL, Ubf + (size_t)l * WL, nullptr,
                      gb + (size_t)(l - 1) * GL, 4 * NHID,
                      nullptr, nullptr, nullptr, s * 64, NHID, ldsA, ldsB);
        } else if (w < 64) {
            const int s = w - 48;    // layer 0: igates == 0 -> fused cell
            gemm_tile(hPrev, Ubf, nullptr, nullptr, 0,
                      c, c, hNext, s * 64, NHID, ldsA, ldsB);
        } else {
            gemm_tile(hPrev + (size_t)3 * HL, Lbf, nullptr,
                      out + (size_t)(t - 1) * BSZ * NOUT, NOUT,
                      nullptr, nullptr, nullptr, (w - 64) * 256, 64, ldsA, ldsB);
        }
        grid_bar(bar, ++epoch * (unsigned)NWG);

        // ---- Phases B/C/D ---------------------------------------------------
        for (int l = 1; l < NLAY; ++l) {
            if (w < 16)
                gemm_tile(hNext + (size_t)(l - 1) * HL, Wbf + (size_t)(l - 1) * WL,
                          gb + (size_t)(l - 1) * GL, nullptr, 0,
                          c + (size_t)l * HL, c + (size_t)l * HL,
                          hNext + (size_t)l * HL, w * 64, NHID, ldsA, ldsB);
            grid_bar(bar, ++epoch * (unsigned)NWG);
        }
        bf16_t* tmp = hPrev; hPrev = hNext; hNext = tmp;
    }
    // ---- final projection: out[127] = h_last[3] @ L^T ----------------------
    if (w >= 64)
        gemm_tile(hPrev + (size_t)3 * HL, Lbf, nullptr,
                  out + (size_t)(STEPS - 1) * BSZ * NOUT, NOUT,
                  nullptr, nullptr, nullptr, (w - 64) * 256, 64, ldsA, ldsB);
}

// ---------------------------------------------------------------------------
__global__ void cvt_f32_bf16(const float* __restrict__ in, bf16_t* __restrict__ out, int n)
{
    for (int i = blockIdx.x * blockDim.x + threadIdx.x; i < n; i += gridDim.x * blockDim.x)
        out[i] = (bf16_t)in[i];
}
__global__ void copy_f32(const float* __restrict__ in, float* __restrict__ out, int n)
{
    for (int i = blockIdx.x * blockDim.x + threadIdx.x; i < n; i += gridDim.x * blockDim.x)
        out[i] = in[i];
}
__global__ void zero_u32(unsigned* p)
{
    if (threadIdx.x == 0 && blockIdx.x == 0) *p = 0u;
}

// ---------------------------------------------------------------------------
extern "C" void kernel_launch(void* const* d_in, const int* in_sizes, int n_in,
                              void* d_out, int out_size, void* d_ws, size_t ws_size,
                              hipStream_t stream)
{
    (void)in_sizes; (void)n_in; (void)out_size; (void)ws_size;

    const float* hx = (const float*)d_in[0];   // [4,64,1024]
    const float* cx = (const float*)d_in[1];   // [4,64,1024]
    const float* W  = (const float*)d_in[2];   // [3,4096,1024]
    const float* U  = (const float*)d_in[3];   // [4,4096,1024]
    const float* L  = (const float*)d_in[4];   // [512,1024]
    // d_in[5] = steps (constant 128 in the reference)

    char* ws = (char*)d_ws;
    size_t off = 0;
    bf16_t*  Wbf = (bf16_t*)(ws + off); off += (size_t)3 * WL * 2;
    bf16_t*  Ubf = (bf16_t*)(ws + off); off += (size_t)4 * WL * 2;
    bf16_t*  Lbf = (bf16_t*)(ws + off); off += (size_t)NOUT * NHID * 2;
    bf16_t*  hA  = (bf16_t*)(ws + off); off += (size_t)4 * HL * 2;
    bf16_t*  hB  = (bf16_t*)(ws + off); off += (size_t)4 * HL * 2;
    float*   c   = (float*) (ws + off); off += (size_t)4 * HL * 4;
    float*   gb  = (float*) (ws + off); off += (size_t)3 * GL * 4;
    unsigned* bar = (unsigned*)(ws + off); off += 256;

    // prepass: weights -> bf16 (L2-resident for the whole recurrence)
    cvt_f32_bf16<<<2048, 256, 0, stream>>>(W,  Wbf, (int)(3 * WL));
    cvt_f32_bf16<<<2048, 256, 0, stream>>>(U,  Ubf, (int)(4 * WL));
    cvt_f32_bf16<<<512,  256, 0, stream>>>(L,  Lbf, NOUT * NHID);
    cvt_f32_bf16<<<512,  256, 0, stream>>>(hx, hA,  4 * HL);
    copy_f32    <<<512,  256, 0, stream>>>(cx, c,   4 * HL);
    zero_u32    <<<1, 64, 0, stream>>>(bar);

    // whole 127-step recurrence (+ all 128 projections) in one launch
    lstm_persistent<<<NWG, 256, 0, stream>>>(
        Wbf, Ubf, Lbf, hA, hB, c, gb, (float*)d_out, bar);
}
